// ExternalMemoryNetwork_9869834846291
// MI455X (gfx1250) — matrix operand order repaired and checked by reading, not compiled, not run on hardware
//
#include <hip/hip_runtime.h>
#include <cstdint>

#define BBATCH 1024
#define MSIZE  200000
#define DDIM   64
#define NTILES (BBATCH / 16)  // 64 batch tiles of 16 rows
#define WPB    4              // waves per block in the flash kernel
#define TILE_WS 1056          // floats per (tile,split) partial: 16 m + 16 l + 16*64 acc

#define VR_STR 72  // row-major V tile stride (bf16 elems), padded vs 64 for banks
#define VT_STR 40  // transposed V tile stride, padded vs 32
#define PT_STR 40  // P tile stride, padded vs 32

typedef __bf16 bf16;
typedef bf16  v16bf __attribute__((ext_vector_type(16)));
typedef bf16  v8bf  __attribute__((ext_vector_type(8)));
typedef float v8f   __attribute__((ext_vector_type(8)));

__device__ inline v8f zero8() {
  v8f z;
#pragma unroll
  for (int i = 0; i < 8; ++i) z[i] = 0.0f;
  return z;
}

__device__ inline v8f wmma_bf16f32(v16bf a, v16bf b, v8f c) {
  // D = A(16x32 bf16) * B(32x16 bf16) + C(16x16 f32)
  return __builtin_amdgcn_wmma_f32_16x16x32_bf16(false, a, false, b, (short)0, c,
                                                 false, false);
}

// A operand (16x32 bf16). Lane l: half h=l>>4, row m=l&15.
//   element e -> K = kOff + 16h + e
__device__ inline v16bf load_A_bf16(const float* __restrict__ src, int row, int half,
                                    int kOff) {
  const float* p0 = src + (size_t)row * DDIM + kOff + 16 * half;
  const float* p1 = p0 + 8;
  v16bf a;
#pragma unroll
  for (int e = 0; e < 8; ++e) {
    a[e]     = (bf16)p0[e];
    a[8 + e] = (bf16)p1[e];
  }
  return a;
}

// B operand (32x16 bf16) from global f32 rows. Lane l: col n; element e -> K=kOff+16h+e.
__device__ inline v16bf load_B_bf16(const float* __restrict__ src, int nrow, int half,
                                    int kOff) {
  const float* p = src + (size_t)nrow * DDIM + kOff + 16 * half;
  v16bf b;
#pragma unroll
  for (int e = 0; e < 16; ++e) b[e] = (bf16)p[e];
  return b;
}

// 16 contiguous bf16 from LDS as two aligned v8bf loads (ds_load_b128 x2)
__device__ inline v16bf lds_b16(const bf16* p) {
  v8bf lo = *(const v8bf*)p;
  v8bf hi = *(const v8bf*)(p + 8);
  v16bf r;
#pragma unroll
  for (int e = 0; e < 8; ++e) {
    r[e]     = lo[e];
    r[8 + e] = hi[e];
  }
  return r;
}

__device__ inline float half_max(float v) {
  v = fmaxf(v, __shfl_xor(v, 1, 32));
  v = fmaxf(v, __shfl_xor(v, 2, 32));
  v = fmaxf(v, __shfl_xor(v, 4, 32));
  v = fmaxf(v, __shfl_xor(v, 8, 32));
  return v;
}

// ---------------------------------------------------------------------------
// Kernel 1: fused flash-attention partial over one memory split.
// 128 threads = 4 waves; waves cooperatively stage each 32x64 V tile into LDS
// (row-major + transposed, double buffered), each wave owns one batch tile.
// ---------------------------------------------------------------------------
__global__ __launch_bounds__(WPB * 32) void emn_flash_partial(
    const float* __restrict__ input, const float* __restrict__ values,
    float* __restrict__ ws, int S, int chunk) {
  __shared__ bf16 vr_s[2][32 * VR_STR];       // [row][col] for score-B operands
  __shared__ bf16 vt_s[2][DDIM * VT_STR];     // [col][row] for PV-B operands
  __shared__ bf16 pt_s[WPB][16 * PT_STR];     // per-wave P tile [row][k]

  const int tid   = threadIdx.x;
  const int lane  = tid & 31;
  const int wave  = tid >> 5;
  const int half  = lane >> 4;
  const int l16   = lane & 15;
  const int tile  = blockIdx.x * WPB + wave;
  const int split = blockIdx.y;
  bf16* pt = pt_s[wave];

  const v16bf a0 = load_A_bf16(input, tile * 16 + l16, half, 0);
  const v16bf a1 = load_A_bf16(input, tile * 16 + l16, half, 32);

  v16bf ones;
#pragma unroll
  for (int e = 0; e < 16; ++e) ones[e] = (bf16)1.0f;

  float mrow[8];
  v8f o[4], oL;
#pragma unroll
  for (int r = 0; r < 8; ++r) mrow[r] = -3.0e38f;
#pragma unroll
  for (int j = 0; j < 4; ++j) o[j] = zero8();
  oL = zero8();

  const int start = split * chunk;
  const int rend  = min(MSIZE, start + chunk);

  int pb = 0;
  for (int base = start; base < rend; base += 32, pb ^= 1) {
    // ---- cooperative V stage: thread loads row=base+lane, cols wave*16..+15 ----
    {
      const int r  = lane;
      const int gr = min(base + r, MSIZE - 1);
      const float* src = values + (size_t)gr * DDIM + wave * 16;
      float f[16];
#pragma unroll
      for (int e = 0; e < 16; ++e) f[e] = src[e];
      v8bf p0, p1;
#pragma unroll
      for (int e = 0; e < 8; ++e) {
        p0[e] = (bf16)f[e];
        p1[e] = (bf16)f[8 + e];
      }
      bf16* dr = &vr_s[pb][r * VR_STR + wave * 16];
      *(v8bf*)dr       = p0;                    // packed ds_store_b128
      *(v8bf*)(dr + 8) = p1;
      bf16* dt = &vt_s[pb][0];
#pragma unroll
      for (int e = 0; e < 8; ++e) {
        dt[(wave * 16 + e) * VT_STR + r]     = p0[e];
        dt[(wave * 16 + 8 + e) * VT_STR + r] = p1[e];
      }
    }
    if (base + 32 < rend)
      __builtin_prefetch((const void*)(values + (size_t)(base + 32) * DDIM + tid * 16),
                         0, 1);
    __syncthreads();  // stage -> consume (double buffer covers the WAR side)

    // ---- score GEMM: S = A(16x64) x V^T(64x32) as 2 subtiles ----
    v8f sc[2];
#pragma unroll
    for (int t = 0; t < 2; ++t) {
      const bf16* p = &vr_s[pb][(t * 16 + l16) * VR_STR + 16 * half];
      v16bf b0 = lds_b16(p);        // K = 0..31
      v16bf b1 = lds_b16(p + 32);   // K = 32..63
      v8f c = zero8();
      c = wmma_bf16f32(a0, b0, c);
      c = wmma_bf16f32(a1, b1, c);
      if (base + t * 16 + l16 >= rend) {
#pragma unroll
        for (int r = 0; r < 8; ++r) c[r] = -3.0e38f;
      }
      sc[t] = c;
    }

    // ---- online softmax: one max/rescale per row over both subtiles ----
#pragma unroll
    for (int r = 0; r < 8; ++r) {
      float s0   = sc[0][r];
      float s1   = sc[1][r];
      float mx   = half_max(fmaxf(s0, s1));
      float mnew = fmaxf(mrow[r], mx);
      float scl  = __expf(mrow[r] - mnew);
      mrow[r]    = mnew;
      float p0   = __expf(s0 - mnew);
      float p1   = __expf(s1 - mnew);
#pragma unroll
      for (int j = 0; j < 4; ++j) o[j][r] *= scl;
      oL[r] *= scl;
      pt[(r + 8 * half) * PT_STR + l16]      = (bf16)p0;
      pt[(r + 8 * half) * PT_STR + 16 + l16] = (bf16)p1;
    }
    asm volatile("s_wait_dscnt 0" ::: "memory");  // cross-lane P-tile RAW
    __builtin_amdgcn_wave_barrier();

    // ---- reload P in A-operand layout ----
    v16bf pa;
    {
      const bf16* q = pt + l16 * PT_STR + 16 * half;
      v8bf qa = *(const v8bf*)q;
      v8bf qb = *(const v8bf*)(q + 8);
#pragma unroll
      for (int e = 0; e < 8; ++e) {
        pa[e]     = qa[e];
        pa[8 + e] = qb[e];
      }
    }
    // ---- accumulate: o[j] += P(16x32) x V(32x16); oL += P x ones (row sums) ----
#pragma unroll
    for (int j = 0; j < 4; ++j) {
      v16bf vb = lds_b16(&vt_s[pb][(16 * j + l16) * VT_STR + 16 * half]);
      o[j] = wmma_bf16f32(pa, vb, o[j]);
    }
    oL = wmma_bf16f32(pa, ones, oL);
  }

  // ---- write per-split partial: [m(16) | l(16) | acc(16x64)] ----
  float* wsb = ws + (size_t)(tile * S + split) * TILE_WS;
  if (l16 == 0) {
#pragma unroll
    for (int r = 0; r < 8; ++r) {
      wsb[r + 8 * half]      = mrow[r];
      wsb[16 + r + 8 * half] = oL[r];  // l accumulated by the ones-WMMA
    }
  }
#pragma unroll
  for (int j = 0; j < 4; ++j)
#pragma unroll
    for (int r = 0; r < 8; ++r)
      wsb[32 + (r + 8 * half) * DDIM + 16 * j + l16] = o[j][r];
}

// ---------------------------------------------------------------------------
// Kernel 2: log-sum-exp merge of the S splits -> retrieved [B,64]
// ---------------------------------------------------------------------------
__global__ __launch_bounds__(DDIM) void emn_flash_reduce(const float* __restrict__ ws,
                                                         float* __restrict__ out,
                                                         int S) {
  const int b    = blockIdx.x;
  const int col  = threadIdx.x;
  const int tile = b >> 4;
  const int rit  = b & 15;

  float gm = -3.0e38f;
  for (int s = 0; s < S; ++s)
    gm = fmaxf(gm, ws[(size_t)(tile * S + s) * TILE_WS + rit]);

  float lsum = 0.0f, acc = 0.0f;
  for (int s = 0; s < S; ++s) {
    const float* p = ws + (size_t)(tile * S + s) * TILE_WS;
    float scl = __expf(p[rit] - gm);
    lsum += p[16 + rit] * scl;
    acc  += p[32 + rit * DDIM + col] * scl;
  }
  out[(size_t)b * DDIM + col] = acc / lsum;
}

// ---------------------------------------------------------------------------
// Kernel 3: read-only flag (all mem_idx == 0 -> skip write path)
// ---------------------------------------------------------------------------
__global__ void emn_flag(const int* __restrict__ mem_idx, int* __restrict__ flag) {
  __shared__ int any;
  if (threadIdx.x == 0) any = 0;
  __syncthreads();
  int local = 0;
  for (int i = threadIdx.x; i < BBATCH; i += blockDim.x) local |= (mem_idx[i] != 0);
  if (local) atomicOr(&any, 1);
  __syncthreads();
  if (threadIdx.x == 0) *flag = any;
}

// ---------------------------------------------------------------------------
// Kernel 4: gated erase/add scatter (gates computed with WMMA as well)
// ---------------------------------------------------------------------------
__global__ __launch_bounds__(32) void emn_write(
    const float* __restrict__ input, const float* __restrict__ values,
    const float* __restrict__ We, const float* __restrict__ be,
    const float* __restrict__ Wa, const float* __restrict__ ba,
    const int* __restrict__ mem_idx, const int* __restrict__ flag,
    float* __restrict__ newvals) {
  if (*flag == 0) return;  // read-only: new_values stays the plain copy

  const int lane = threadIdx.x & 31;
  const int half = lane >> 4;
  const int l16  = lane & 15;
  const int tile = blockIdx.x;

  const v16bf a0 = load_A_bf16(input, tile * 16 + l16, half, 0);
  const v16bf a1 = load_A_bf16(input, tile * 16 + l16, half, 32);

  int mi[8];
#pragma unroll
  for (int r = 0; r < 8; ++r) mi[r] = mem_idx[tile * 16 + r + 8 * half];

#pragma unroll
  for (int j = 0; j < 4; ++j) {
    const int ncol = 16 * j + l16;
    v8f ce = zero8(), ca = zero8();
    ce = wmma_bf16f32(a0, load_B_bf16(We, ncol, half, 0), ce);
    ce = wmma_bf16f32(a1, load_B_bf16(We, ncol, half, 32), ce);
    ca = wmma_bf16f32(a0, load_B_bf16(Wa, ncol, half, 0), ca);
    ca = wmma_bf16f32(a1, load_B_bf16(Wa, ncol, half, 32), ca);
    const float bi_e = be[ncol];
    const float bi_a = ba[ncol];
#pragma unroll
    for (int r = 0; r < 8; ++r) {
      float ew  = 1.0f / (1.0f + __expf(-(ce[r] + bi_e)));
      float aw  = tanhf(ca[r] + bi_a);
      float old = values[(size_t)mi[r] * DDIM + ncol];
      newvals[(size_t)mi[r] * DDIM + ncol] = (1.0f - ew) * old + aw;
    }
  }
}

// ---------------------------------------------------------------------------
extern "C" void kernel_launch(void* const* d_in, const int* in_sizes, int n_in,
                              void* d_out, int out_size, void* d_ws, size_t ws_size,
                              hipStream_t stream) {
  (void)in_sizes; (void)n_in; (void)out_size;

  const int*   mem_idx = (const int*)d_in[0];
  const float* input   = (const float*)d_in[1];
  const float* values  = (const float*)d_in[2];
  const float* We      = (const float*)d_in[3];
  const float* beb     = (const float*)d_in[4];
  const float* Wa      = (const float*)d_in[5];
  const float* bab     = (const float*)d_in[6];

  float* retrieved = (float*)d_out;
  float* newvals   = retrieved + (size_t)BBATCH * DDIM;

  int*   flag = (int*)d_ws;
  float* ws   = (float*)((char*)d_ws + 256);

  const size_t perSplit = (size_t)NTILES * TILE_WS * sizeof(float);  // ~270 KB
  const size_t avail    = ws_size > 256 ? ws_size - 256 : 0;
  int S = (int)(avail / perSplit);
  if (S < 1) S = 1;
  if (S > 64) S = 64;
  const int chunk = (((MSIZE + S - 1) / S) + 31) & ~31;  // 32-row aligned splits

  emn_flag<<<1, 256, 0, stream>>>(mem_idx, flag);

  dim3 g1(NTILES / WPB, S);
  emn_flash_partial<<<g1, WPB * 32, 0, stream>>>(input, values, ws, S, chunk);

  emn_flash_reduce<<<BBATCH, DDIM, 0, stream>>>(ws, retrieved, S);

  hipMemcpyAsync(newvals, values, (size_t)MSIZE * DDIM * sizeof(float),
                 hipMemcpyDeviceToDevice, stream);

  emn_write<<<NTILES, 32, 0, stream>>>(input, values, We, beb, Wa, bab, mem_idx, flag,
                                       newvals);
}
